// SelfAttentionWide_54941221651255
// MI455X (gfx1250) — matrix operand order repaired and checked by reading, not compiled, not run on hardware
//
#include <hip/hip_runtime.h>
#include <cstdint>

typedef unsigned short u16;
typedef __attribute__((ext_vector_type(16))) __bf16 v16bf;
typedef __attribute__((ext_vector_type(8)))  float  v8f;

constexpr int BATCH = 4;
constexpr int SEQ   = 4096;
constexpr int EMB   = 64;
constexpr int WIDE  = 512;
// 1/sqrt(512) * log2(e): scores are produced directly in log2-domain so the
// softmax can use v_exp_f32 (exp2) without the per-element *log2e multiply.
constexpr float SCALE_LOG2E = 0.04419417382415922f * 1.4426950408889634f;

union Frag16 { uint4 u4[2]; unsigned u[8]; v16bf v; };

__device__ __forceinline__ u16 f2bf(float f) {
  union { float f; unsigned u; } x{f};
  unsigned r = x.u + 0x7FFFu + ((x.u >> 16) & 1u);   // round-to-nearest-even
  return (u16)(r >> 16);
}
__device__ __forceinline__ float bf2f(u16 h) {
  union { unsigned u; float f; } x{((unsigned)h) << 16};
  return x.f;
}
// pack two f32 -> bf16x2 dword: one add per value + one v_perm_b32
__device__ __forceinline__ unsigned pack_bf16(float lo, float hi) {
  union { float f; unsigned u; } a{lo}, b{hi};
  return __builtin_amdgcn_perm(b.u + 0x8000u, a.u + 0x8000u, 0x07060302u);
}
__device__ __forceinline__ v8f wmma_bf16(v16bf a, v16bf b, v8f c) {
  // D(16x16,f32) = A(16x32,bf16) x B(32x16,bf16) + C
  return __builtin_amdgcn_wmma_f32_16x16x32_bf16(false, a, false, b, (short)0, c, false, false);
}

// ---------------------------------------------------------------------------
// Kernel 1: M = (Wq Wk^T)*log2e/sqrt(512), P = Wv Wm -- both 64x64, bf16
// ---------------------------------------------------------------------------
__global__ __launch_bounds__(256) void prep_weights_k(
    const float* __restrict__ Wq, const float* __restrict__ Wk,
    const float* __restrict__ Wv, const float* __restrict__ Wm,
    u16* __restrict__ Mb, u16* __restrict__ Pb) {
  const int tid = threadIdx.x;
  for (int i = 0; i < 16; ++i) {
    const int idx = i * 256 + tid;          // 4096 outputs per matrix
    const int e = idx >> 6, c = idx & 63;
    float am = 0.f, ap = 0.f;
    for (int w = 0; w < WIDE; ++w) {
      am += Wq[e * WIDE + w] * Wk[c * WIDE + w];   // (WqWk^T)[e][c]
      ap += Wv[e * WIDE + w] * Wm[w * EMB + c];    // (WvWm)[e][c]
    }
    Mb[idx] = f2bf(am * SCALE_LOG2E);
    Pb[idx] = f2bf(ap);
  }
}

// ---------------------------------------------------------------------------
// Kernel 2: Q' = x*M (bf16 [B*S][64]), V'^T = (x*P)^T (bf16 [B][64][S]),
//           Xb = bf16(x) ([B*S][64]).  16 rows per 256-thread block.
// ---------------------------------------------------------------------------
__global__ __launch_bounds__(256) void proj_k(
    const float* __restrict__ x, const u16* __restrict__ Mb, const u16* __restrict__ Pb,
    u16* __restrict__ Qb, u16* __restrict__ Xb, u16* __restrict__ VbT) {
  __shared__ __align__(16) float xls[16][64];
  __shared__ __align__(16) u16   mls[EMB * EMB];
  __shared__ __align__(16) u16   pls[EMB * EMB];
  const int tid = threadIdx.x;
#pragma unroll
  for (int i = 0; i < 2; ++i) {             // 512 uint4 per 64x64 bf16 matrix
    const int c = i * 256 + tid;
    ((uint4*)mls)[c] = ((const uint4*)Mb)[c];
    ((uint4*)pls)[c] = ((const uint4*)Pb)[c];
  }
  const int row = tid >> 4, cg = tid & 15;
  const long grow = (long)blockIdx.x * 16 + row;
  const float4 xv = *(const float4*)&x[grow * 64 + cg * 4];
  *(float4*)&xls[row][cg * 4] = xv;
  Xb[grow * 64 + cg * 4 + 0] = f2bf(xv.x);
  Xb[grow * 64 + cg * 4 + 1] = f2bf(xv.y);
  Xb[grow * 64 + cg * 4 + 2] = f2bf(xv.z);
  Xb[grow * 64 + cg * 4 + 3] = f2bf(xv.w);
  __syncthreads();
  const int col0 = cg * 4;
  float aq[4] = {0, 0, 0, 0}, av[4] = {0, 0, 0, 0};
  for (int e = 0; e < 64; ++e) {
    const float xe = xls[row][e];
#pragma unroll
    for (int i = 0; i < 4; ++i) {
      aq[i] += xe * bf2f(mls[e * 64 + col0 + i]);
      av[i] += xe * bf2f(pls[e * 64 + col0 + i]);
    }
  }
  const int bb = (int)(grow >> 12), ss = (int)(grow & (SEQ - 1));
#pragma unroll
  for (int i = 0; i < 4; ++i) {
    Qb[grow * 64 + col0 + i] = f2bf(aq[i]);
    VbT[((long)(bb * 64 + col0 + i)) * SEQ + ss] = f2bf(av[i]);  // transposed
  }
}

// ---------------------------------------------------------------------------
// Kernel 3: flash attention, transposed formulation:
//   S^T = Xk x Q'^T   (keys in VGPR rows, queries in lanes)
//   O^T += V'^T x P^T (out-dims in VGPR rows, queries in lanes)
// Softmax stats are per-lane scalars; 6 cross-lane ops per 32-key tile.
// 128 threads = 4 wave32; wave owns 16 queries; WG owns 64 queries.
// ---------------------------------------------------------------------------
__global__ __launch_bounds__(128) void fa_k(
    const u16* __restrict__ Qb, const u16* __restrict__ Xb,
    const u16* __restrict__ VbT, float* __restrict__ out) {
  constexpr int KT = 32;
  constexpr int XSTR = 72;   // bf16 row stride, conflict-free b128 reads
  constexpr int VSTR = 40;
  __shared__ __align__(16) u16 xs [2][KT ][XSTR];   // keys: [key][e]
  __shared__ __align__(16) u16 vts[2][EMB][VSTR];   // V'^T: [d][key]

  const int tid  = threadIdx.x;
  const int lane = tid & 31, wv = tid >> 5;
  const int hh = lane >> 4, ln = lane & 15;
  const long q0 = (long)blockIdx.x * 64;           // flat token index of WG's queries
  const int  b  = (int)(q0 >> 12);                 // batch
  const long kbrow = (long)b * SEQ;                // Xb row base for this batch

  // Persistent B-fragments of Q'^T (two e-blocks of 32).
  // B layout: lane = query col; lanes 0-15 hold K 0..15, lanes 16-31 hold K 16..31.
  Frag16 bq0, bq1;
  {
    const u16* qrow = Qb + (q0 + wv * 16 + ln) * 64;
    bq0.u4[0] = *(const uint4*)(qrow + hh * 16);
    bq0.u4[1] = *(const uint4*)(qrow + hh * 16 + 8);
    bq1.u4[0] = *(const uint4*)(qrow + 32 + hh * 16);
    bq1.u4[1] = *(const uint4*)(qrow + 32 + hh * 16 + 8);
  }

  v8f oacc[4];
#pragma unroll
  for (int t4 = 0; t4 < 4; ++t4) oacc[t4] = (v8f){0, 0, 0, 0, 0, 0, 0, 0};
  float mrun = -1e30f, lrun = 0.f;                 // per-lane (per-query) scalars

  auto stage = [&](int buf, int key0) {
#pragma unroll
    for (int j = 0; j < 2; ++j) {                  // key tile: 32 rows x 64 bf16
      const int id = tid + 128 * j;
      const int kr = id >> 3, c = (id & 7) * 8;
      *(uint4*)&xs[buf][kr][c] = *(const uint4*)&Xb[(kbrow + key0 + kr) * 64 + c];
    }
#pragma unroll
    for (int j = 0; j < 2; ++j) {                  // V'^T tile: 64 rows x 32 keys
      const int id = tid + 128 * j;
      const int d = id >> 2, c = (id & 3) * 8;
      *(uint4*)&vts[buf][d][c] =
          *(const uint4*)&VbT[((long)(b * 64 + d)) * SEQ + key0 + c];
    }
  };

  stage(0, 0);
  const int nT = SEQ / KT;
  for (int t = 0; t < nT; ++t) {
    const int buf = t & 1;
    __syncthreads();                               // tile `buf` ready; prev reads done
    if (t + 1 < nT) stage(buf ^ 1, (t + 1) * KT);  // overlap next stage with compute

    // ---- hoist ALL score A-fragments (8x ds_load_b128), then 4 WMMAs:
    // amortizes s_wait_dscnt across the WMMA burst.
    // A layout: lane<16 -> key row=lane, e 0-15; lane>=16 -> e 16-31.
    Frag16 ax[2][2];
#pragma unroll
    for (int kt = 0; kt < 2; ++kt) {
      const u16* krow = &xs[buf][kt * 16 + ln][0];
      ax[kt][0].u4[0] = *(const uint4*)(krow + hh * 16);
      ax[kt][0].u4[1] = *(const uint4*)(krow + hh * 16 + 8);
      ax[kt][1].u4[0] = *(const uint4*)(krow + 32 + hh * 16);
      ax[kt][1].u4[1] = *(const uint4*)(krow + 32 + hh * 16 + 8);
    }
    v8f sc[2];
#pragma unroll
    for (int kt = 0; kt < 2; ++kt) {
      v8f c = (v8f){0, 0, 0, 0, 0, 0, 0, 0};
      c = wmma_bf16(ax[kt][0].v, bq0.v, c);
      c = wmma_bf16(ax[kt][1].v, bq1.v, c);
      sc[kt] = c;                                  // lane: 16 keys (log2-domain)
    }

    // ---- issue V'^T A-fragment loads NOW; latency hides under softmax VALU.
    Frag16 av[4];
#pragma unroll
    for (int t4 = 0; t4 < 4; ++t4) {
      const u16* vrow = &vts[buf][t4 * 16 + ln][0];
      av[t4].u4[0] = *(const uint4*)(vrow + hh * 16);
      av[t4].u4[1] = *(const uint4*)(vrow + hh * 16 + 8);
    }

    // ---- online softmax in log2-domain: in-lane over 16 keys + 1 xor-16/stat
    float mx = fmaxf(sc[0][0], sc[1][0]);
#pragma unroll
    for (int r = 1; r < 8; ++r) mx = fmaxf(mx, fmaxf(sc[0][r], sc[1][r]));
    mx = fmaxf(mx, __shfl_xor(mx, 16, 32));        // other 16 keys (partner half)
    const float mn = fmaxf(mrun, mx);
    const float al = __builtin_amdgcn_exp2f(mrun - mn);
    mrun = mn;

    float p[2][8];
    float sum = 0.f;
#pragma unroll
    for (int t2 = 0; t2 < 2; ++t2)
#pragma unroll
      for (int r = 0; r < 8; ++r) {
        const float e = __builtin_amdgcn_exp2f(sc[t2][r] - mn);
        p[t2][r] = e;
        sum += e;
      }
    sum += __shfl_xor(sum, 16, 32);
    lrun = lrun * al + sum;
#pragma unroll
    for (int t4 = 0; t4 < 4; ++t4) oacc[t4] *= al; // vector * per-lane scalar

    // ---- build P^T B-fragment in registers: pack pairs (v_perm) and swap one
    // half-tile with the xor-16 partner lane.
    unsigned pk[2][4];
#pragma unroll
    for (int t2 = 0; t2 < 2; ++t2)
#pragma unroll
      for (int j = 0; j < 4; ++j)
        pk[t2][j] = pack_bf16(p[t2][2 * j], p[t2][2 * j + 1]);
    Frag16 bp;
#pragma unroll
    for (int j = 0; j < 4; ++j) {
      const unsigned send = hh ? pk[0][j] : pk[1][j];   // what partner needs
      const unsigned recv = __shfl_xor(send, 16, 32);
      bp.u[j]     = hh ? recv     : pk[0][j];           // keys hh*16 + 0..7
      bp.u[4 + j] = hh ? pk[1][j] : recv;               // keys hh*16 + 8..15
    }

    // ---- O^T += V'^T(64x32) @ P^T(32x16): 4 independent WMMAs
#pragma unroll
    for (int t4 = 0; t4 < 4; ++t4)
      oacc[t4] = wmma_bf16(av[t4].v, bp.v, oacc[t4]);
  }

  // ---- epilogue: out[q][d] = O^T[d][q] / l(q)   (fp32 [B,S,64])
  const float inv = 1.f / lrun;
  const long qtok = q0 + wv * 16 + ln;
#pragma unroll
  for (int t4 = 0; t4 < 4; ++t4)
#pragma unroll
    for (int r = 0; r < 8; ++r) {
      const int d = t4 * 16 + hh * 8 + r;
      out[qtok * 64 + d] = oacc[t4][r] * inv;
    }
}

// ---------------------------------------------------------------------------
extern "C" void kernel_launch(void* const* d_in, const int* in_sizes, int n_in,
                              void* d_out, int out_size, void* d_ws, size_t ws_size,
                              hipStream_t stream) {
  (void)in_sizes; (void)n_in; (void)out_size; (void)ws_size;
  const float* x  = (const float*)d_in[0];
  const float* Wq = (const float*)d_in[1];
  const float* Wk = (const float*)d_in[2];
  const float* Wv = (const float*)d_in[3];
  const float* Wm = (const float*)d_in[4];
  float* out = (float*)d_out;

  // workspace layout (bf16 buffers):
  //   Mb 64*64 @0 ; Pb 64*64 @8KiB ; Qb [B*S][64] @16KiB ;
  //   Xb [B*S][64] @16KiB+2MiB ; VbT [B][64][S] @16KiB+4MiB
  char* ws = (char*)d_ws;
  constexpr size_t TOK = (size_t)BATCH * SEQ * EMB * sizeof(u16);  // 2 MiB
  u16* Mb  = (u16*)(ws);
  u16* Pb  = (u16*)(ws + 8192);
  u16* Qb  = (u16*)(ws + 16384);
  u16* Xb  = (u16*)(ws + 16384 + TOK);
  u16* VbT = (u16*)(ws + 16384 + 2 * TOK);

  prep_weights_k<<<1, 256, 0, stream>>>(Wq, Wk, Wv, Wm, Mb, Pb);
  proj_k<<<(BATCH * SEQ) / 16, 256, 0, stream>>>(x, Mb, Pb, Qb, Xb, VbT);
  fa_k<<<(BATCH * SEQ) / 64, 128, 0, stream>>>(Qb, Xb, VbT, out);
}
